// RCMoEQNetwork_47330539602633
// MI455X (gfx1250) — compile-verified, matrix operand order checked by reading, not run on hardware
//
#include <hip/hip_runtime.h>

// Problem constants (match reference)
#define OBS 256
#define ACT 64
#define HID 512
#define NE  8
#define GH  64
#define NB  65536

typedef __attribute__((ext_vector_type(16))) __bf16 v16bf;
typedef __attribute__((ext_vector_type(8)))  float  v8f;
typedef __attribute__((ext_vector_type(4)))  unsigned int u32x4;

union FragU { v16bf bf; u32x4 q[2]; };

__device__ __forceinline__ unsigned short f2bf(float x) {
  unsigned int u = __float_as_uint(x);
  unsigned int r = u + 0x7FFFu + ((u >> 16) & 1u);   // round-to-nearest-even
  return (unsigned short)(r >> 16);
}

__device__ __forceinline__ float silu_f(float v) {
  return v / (1.0f + __expf(-v));
}

// 16x32 bf16 fragment load (row-major source, 16 rows x stride cols).
// ISA 16-bit A layout: lane m = lane&15, kb = (lane>>4)*8;
// VGPRs 0-3 hold K=kb..kb+7, VGPRs 4-7 hold K=kb+16..kb+23.
// Pre-transposed B uses the identical pattern (lane = output column).
__device__ __forceinline__ v16bf load_frag(const unsigned short* base, int stride, int lane) {
  const int r  = lane & 15;
  const int kb = (lane >> 4) << 3;
  const unsigned short* p = base + r * stride + kb;
  FragU f;
  f.q[0] = *(const u32x4*)(p);
  f.q[1] = *(const u32x4*)(p + 16);
  return f.bf;
}

__device__ __forceinline__ v8f wmma_bf16(v16bf a, v16bf b, v8f c) {
  return __builtin_amdgcn_wmma_f32_16x16x32_bf16(false, a, false, b, (short)0, c, false, false);
}

#define VZERO ((v8f){0,0,0,0,0,0,0,0})

// 32xK @ Kx32 pass: two M-tiles share two B fragments -> 4 independent WMMA chains,
// and each B tile (the bandwidth-critical operand) feeds 2 WMMAs.
template<int KT>
__device__ __forceinline__ void gemm_quad(const unsigned short* A, int strideA,
                                          const unsigned short* B, int strideB, int n0,
                                          v8f& ca0, v8f& ca1, v8f& cb0, v8f& cb1, int lane) {
#pragma unroll
  for (int kt = 0; kt < KT; ++kt) {
    v16bf aa = load_frag(A + kt * 32,               strideA, lane);   // rows 0..15
    v16bf ab = load_frag(A + 16 * strideA + kt * 32, strideA, lane);  // rows 16..31
    v16bf b0 = load_frag(B + (size_t)n0        * strideB + kt * 32, strideB, lane);
    v16bf b1 = load_frag(B + (size_t)(n0 + 16) * strideB + kt * 32, strideB, lane);
    ca0 = wmma_bf16(aa, b0, ca0);
    ca1 = wmma_bf16(aa, b1, ca1);
    cb0 = wmma_bf16(ab, b0, cb0);
    cb1 = wmma_bf16(ab, b1, cb1);
  }
}

// bias + silu + bf16 store of a 16x32 output pair into LDS hidden buffer
__device__ __forceinline__ void act_store(v8f c0, v8f c1, const float* bias, int n0,
                                          unsigned short* h, int stride, int lane) {
  int col = n0 + (lane & 15);
  float bv0 = bias[col], bv1 = bias[col + 16];
#pragma unroll
  for (int r = 0; r < 8; ++r) {
    int m = r + ((lane >> 4) << 3);             // C/D layout: VGPR r -> rows r / r+8
    h[m * stride + col]      = f2bf(silu_f(c0[r] + bv0));
    h[m * stride + col + 16] = f2bf(silu_f(c1[r] + bv1));
  }
}

// ---------------- weight convert+transpose (f32 -> bf16, N-major) ----------------
#define SZ_WG1T (GH * OBS)                 // 16384
#define OFF_W1T (SZ_WG1T)
#define SZ_W1T  (NE * HID * OBS)           // 1048576
#define OFF_W2T (OFF_W1T + SZ_W1T)
#define SZ_W2T  (NE * HID * HID)           // 2097152
#define OFF_W3T (OFF_W2T + SZ_W2T)
#define SZ_W3T  (NE * ACT * HID)           // 262144
#define CVT_TOTAL (OFF_W3T + SZ_W3T)       // 3424256 elems = 6.85 MB bf16 (L2-resident)

__global__ void convert_weights(const float* __restrict__ Wg1, const float* __restrict__ W1,
                                const float* __restrict__ W2,  const float* __restrict__ W3,
                                unsigned short* __restrict__ ws) {
  int i = blockIdx.x * blockDim.x + threadIdx.x;
  if (i >= CVT_TOTAL) return;
  float v;
  if (i < OFF_W1T) {                         // Wg1T[n][k] <- Wg1[k][n], 64x256
    int n = i >> 8, k = i & 255;
    v = Wg1[k * GH + n];
  } else if (i < OFF_W2T) {                  // W1T[e][n][k] <- W1[e][k][n], 512x256
    int j = i - OFF_W1T;
    int e = j >> 17, t = j & 131071;
    int n = t >> 8, k = t & 255;
    v = W1[e * (OBS * HID) + k * HID + n];
  } else if (i < OFF_W3T) {                  // W2T[e][n][k] <- W2[e][k][n], 512x512
    int j = i - OFF_W2T;
    int e = j >> 18, t = j & 262143;
    int n = t >> 9, k = t & 511;
    v = W2[e * (HID * HID) + k * HID + n];
  } else {                                   // W3T[e][n][k] <- W3[e][k][n], 64x512
    int j = i - OFF_W3T;
    int e = j >> 15, t = j & 32767;
    int n = t >> 9, k = t & 511;
    v = W3[e * (HID * ACT) + k * ACT + n];
  }
  ws[i] = f2bf(v);
}

// ---------------- fused MoE kernel: 2 waves/block, 32 rows per wave ----------------
__global__ __launch_bounds__(64) void moe_fused(
    const float* __restrict__ x,   const float* __restrict__ bg1,
    const float* __restrict__ Wg2, const float* __restrict__ bg2,
    const float* __restrict__ b1,  const float* __restrict__ b2, const float* __restrict__ b3,
    const unsigned short* __restrict__ Wg1T, const unsigned short* __restrict__ W1T,
    const unsigned short* __restrict__ W2T,  const unsigned short* __restrict__ W3T,
    float* __restrict__ out_q, float* __restrict__ out_w)
{
  // 160 KB/block -> 2 blocks per 320 KB WGP
  __shared__ alignas(16) unsigned short xs [2][32 * OBS];   // 32 KB
  __shared__ alignas(16) unsigned short h1s[2][32 * HID];   // 64 KB
  __shared__ alignas(16) unsigned short h2s[2][32 * HID];   // 64 KB

  const int wave = threadIdx.x >> 5;
  const int lane = threadIdx.x & 31;
  const int row0 = blockIdx.x * 64 + wave * 32;   // this wave's 32 batch rows

  unsigned short* xsw = &xs [wave][0];
  unsigned short* h1w = &h1s[wave][0];
  unsigned short* h2w = &h2s[wave][0];
  float* gsm = (float*)h2w;                       // 32x64 gate overlay (dead before L2)

  // ---- stage x tile (32 x 256) f32 global -> bf16 LDS (wave-private, no barrier) ----
  for (int i = lane; i < (32 * OBS) / 4; i += 32) {
    int r = i >> 6;                // 64 float4 per row
    int c = (i & 63) << 2;
    const float4 v = *(const float4*)(x + (size_t)(row0 + r) * OBS + c);
    ushort4 o;
    o.x = f2bf(v.x); o.y = f2bf(v.y); o.z = f2bf(v.z); o.w = f2bf(v.w);
    *(ushort4*)(xsw + r * OBS + c) = o;
  }

  // ---- gate hidden: g = silu(x @ Wg1 + bg1), 32x64 via WMMA ----
#pragma unroll
  for (int t = 0; t < 2; ++t) {
    int n0 = t * 32;
    v8f ca0 = VZERO, ca1 = VZERO, cb0 = VZERO, cb1 = VZERO;
    gemm_quad<OBS / 32>(xsw, OBS, Wg1T, OBS, n0, ca0, ca1, cb0, cb1, lane);
    int col = n0 + (lane & 15);
    float bv0 = bg1[col], bv1 = bg1[col + 16];
#pragma unroll
    for (int r = 0; r < 8; ++r) {
      int m = r + ((lane >> 4) << 3);
      gsm[m * GH + col]             = silu_f(ca0[r] + bv0);
      gsm[m * GH + col + 16]        = silu_f(ca1[r] + bv1);
      gsm[(m + 16) * GH + col]      = silu_f(cb0[r] + bv0);
      gsm[(m + 16) * GH + col + 16] = silu_f(cb1[r] + bv1);
    }
  }

  // ---- gate logits + softmax (f32 scalar; every lane owns one of the 32 rows) ----
  {
    int r = lane;
    float lg[NE];
#pragma unroll
    for (int e = 0; e < NE; ++e) lg[e] = bg2[e];
    for (int k = 0; k < GH; ++k) {
      float gv = gsm[r * GH + k];
#pragma unroll
      for (int e = 0; e < NE; ++e) lg[e] = fmaf(gv, Wg2[k * NE + e], lg[e]);
    }
    float mx = lg[0];
#pragma unroll
    for (int e = 1; e < NE; ++e) mx = fmaxf(mx, lg[e]);
    float s = 0.f;
#pragma unroll
    for (int e = 0; e < NE; ++e) { lg[e] = __expf(lg[e] - mx); s += lg[e]; }
    float inv = 1.f / s;
#pragma unroll
    for (int e = 0; e < NE; ++e)
      out_w[(size_t)(row0 + r) * NE + e] = lg[e] * inv;   // output #2 (re-read for mixing)
  }

  // ---- experts ----
  v8f mix[8];                                   // [mtile 0/1][colgroup 0..3]
#pragma unroll
  for (int t = 0; t < 8; ++t) mix[t] = VZERO;

  for (int e = 0; e < NE; ++e) {
    // L1: h1 = silu(x @ W1[e] + b1[e])   (32x256 @ 256x512)
    const unsigned short* w1 = W1T + (size_t)e * (HID * OBS);
    const float* bb1 = b1 + e * HID;
    for (int n0 = 0; n0 < HID; n0 += 32) {
      v8f ca0 = VZERO, ca1 = VZERO, cb0 = VZERO, cb1 = VZERO;
      gemm_quad<OBS / 32>(xsw, OBS, w1, OBS, n0, ca0, ca1, cb0, cb1, lane);
      act_store(ca0, ca1, bb1, n0, h1w,            HID, lane);
      act_store(cb0, cb1, bb1, n0, h1w + 16 * HID, HID, lane);
    }
    // L2: h2 = silu(h1 @ W2[e] + b2[e])  (32x512 @ 512x512)
    const unsigned short* w2 = W2T + (size_t)e * (HID * HID);
    const float* bb2 = b2 + e * HID;
    for (int n0 = 0; n0 < HID; n0 += 32) {
      v8f ca0 = VZERO, ca1 = VZERO, cb0 = VZERO, cb1 = VZERO;
      gemm_quad<HID / 32>(h1w, HID, w2, HID, n0, ca0, ca1, cb0, cb1, lane);
      act_store(ca0, ca1, bb2, n0, h2w,            HID, lane);
      act_store(cb0, cb1, bb2, n0, h2w + 16 * HID, HID, lane);
    }
    // L3: q = h2 @ W3[e] + b3[e]; mixed += weights[:,e] * q   (32x512 @ 512x64)
    const unsigned short* w3 = W3T + (size_t)e * (ACT * HID);
#pragma unroll
    for (int t = 0; t < 2; ++t) {
      int n0 = t * 32;
      v8f ca0 = VZERO, ca1 = VZERO, cb0 = VZERO, cb1 = VZERO;
      gemm_quad<HID / 32>(h2w, HID, w3, HID, n0, ca0, ca1, cb0, cb1, lane);
      int col = n0 + (lane & 15);
      float bv0 = b3[e * ACT + col], bv1 = b3[e * ACT + col + 16];
#pragma unroll
      for (int r = 0; r < 8; ++r) {
        int m = r + ((lane >> 4) << 3);
        float wa = out_w[(size_t)(row0 + m) * NE + e];        // same-wave store->load, HW-ordered
        float wb = out_w[(size_t)(row0 + 16 + m) * NE + e];
        mix[0 + 2 * t][r]     += wa * (ca0[r] + bv0);
        mix[0 + 2 * t + 1][r] += wa * (ca1[r] + bv1);
        mix[4 + 2 * t][r]     += wb * (cb0[r] + bv0);
        mix[4 + 2 * t + 1][r] += wb * (cb1[r] + bv1);
      }
    }
  }

  // ---- write mixed_q (32 x 64) ----
#pragma unroll
  for (int mt = 0; mt < 2; ++mt) {
#pragma unroll
    for (int t = 0; t < 4; ++t) {
      int col = t * 16 + (lane & 15);
#pragma unroll
      for (int r = 0; r < 8; ++r) {
        int m = r + ((lane >> 4) << 3) + mt * 16;
        out_q[(size_t)(row0 + m) * ACT + col] = mix[mt * 4 + t][r];
      }
    }
  }
}

extern "C" void kernel_launch(void* const* d_in, const int* in_sizes, int n_in,
                              void* d_out, int out_size, void* d_ws, size_t ws_size,
                              hipStream_t stream) {
  (void)in_sizes; (void)n_in; (void)out_size; (void)ws_size;
  const float* x   = (const float*)d_in[0];
  const float* Wg1 = (const float*)d_in[1];
  const float* bg1 = (const float*)d_in[2];
  const float* Wg2 = (const float*)d_in[3];
  const float* bg2 = (const float*)d_in[4];
  const float* W1  = (const float*)d_in[5];
  const float* b1  = (const float*)d_in[6];
  const float* W2  = (const float*)d_in[7];
  const float* b2  = (const float*)d_in[8];
  const float* W3  = (const float*)d_in[9];
  const float* b3  = (const float*)d_in[10];

  unsigned short* wsb = (unsigned short*)d_ws;
  float* out_q = (float*)d_out;
  float* out_w = out_q + (size_t)NB * ACT;

  convert_weights<<<(CVT_TOTAL + 255) / 256, 256, 0, stream>>>(Wg1, W1, W2, W3, wsb);
  moe_fused<<<NB / 64, 64, 0, stream>>>(x, bg1, Wg2, bg2, b1, b2, b3,
                                        wsb /*Wg1T*/, wsb + OFF_W1T, wsb + OFF_W2T, wsb + OFF_W3T,
                                        out_q, out_w);
}
